// MultiHeadAttentionLayer_27693949124626
// MI455X (gfx1250) — matrix operand order repaired and checked by reading, not compiled, not run on hardware
//
#include <hip/hip_runtime.h>
#include <hip/hip_bf16.h>
#include <math.h>

// ---------------------------------------------------------------------------
// Transformer encoder layer for MI455X (gfx1250, wave32, WMMA 16x16x32 f16).
// B=4, N=2048, D=512, H=8, FF=2048, DK=64.
// ---------------------------------------------------------------------------

typedef __attribute__((ext_vector_type(16))) _Float16 v16h;
typedef __attribute__((ext_vector_type(8)))  _Float16 v8h;
typedef __attribute__((ext_vector_type(8)))  float    v8f;

#define BB   4
#define NNq  2048
#define DDm  512
#define HHn  8
#define FFn  2048
#define DKK  64
#define BNt  (BB * NNq)      // 8192 tokens
#define EPSb 1e-5f

// ---------------- WMMA fragment helpers (wave32 layouts, CDNA5 ISA 7.12.2) --
//
// A (16x32 f16, row-major source [M,K], per lane = row M=lane%16):
//   elements 0..7  <-> K = 8*(lane/16) + 0..7        (contiguous 16B)
//   elements 8..15 <-> K = 16 + 8*(lane/16) + 0..7   (contiguous 16B)
static __device__ __forceinline__ v16h load_afrag(const _Float16* __restrict__ A,
                                                  long rowBase, int lda, int kBase,
                                                  int lane) {
    int row = lane & 15, half = lane >> 4;
    const _Float16* p = A + (rowBase + row) * (long)lda + kBase + 8 * half;
    v8h lo = *(const v8h*)p;
    v8h hi = *(const v8h*)(p + 16);
    v16h a;
#pragma unroll
    for (int i = 0; i < 8; ++i) { a[i] = lo[i]; a[i + 8] = hi[i]; }
    return a;
}

// B (32x16 f16): source is B^T row-major [N,K]; per lane = column N=lane%16:
//   elements 0..7  <-> Kdim = 8*(lane/16) + 0..7, elements 8..15 <-> Kdim = 16 + 8*(lane/16) + 0..7
static __device__ __forceinline__ v16h load_bfrag(const _Float16* __restrict__ Bt,
                                                  long colBase, long ldb, long kBase,
                                                  int lane) {
    int col = lane & 15, half = lane >> 4;
    const _Float16* p = Bt + (colBase + col) * ldb + kBase + 8 * half;
    v8h lo = *(const v8h*)p;
    v8h hi = *(const v8h*)(p + 16);
    v16h b;
#pragma unroll
    for (int i = 0; i < 8; ++i) { b[i] = lo[i]; b[i + 8] = hi[i]; }
    return b;
}

static __device__ __forceinline__ v8f vzero8() {
    v8f z = {0.f, 0.f, 0.f, 0.f, 0.f, 0.f, 0.f, 0.f};
    return z;
}

static __device__ __forceinline__ v8f wmma_f16(v16h a, v16h b, v8f c) {
    return __builtin_amdgcn_wmma_f32_16x16x32_f16(false, a, false, b,
                                                  (short)0, c, false, false);
}

// ---------------- Generic WMMA GEMM ----------------------------------------
// C[M,Nc] = scale * (A[M,K] @ B[K,Nc]) + bias[col]  (B given as Bt[Nc,K] f16)
// optional relu; optional residual add (f32, same shape); f32 and/or f16 out.
// Each wave owns RT 16-row tiles x 64 cols; block = 4 waves.
// grid = (M/(64*RT), Nc/64, batch).
// K-loop is software-pipelined: iteration k+1's fragments are issued before
// iteration k's WMMAs so s_wait_loadcnt overlaps with in-flight loads.
// launch_bounds(...,1): allow full VGPR file, no scratch spills.
template <int RT>
__global__ void __launch_bounds__(128, 1)
gemm_wmma_kernel(const _Float16* __restrict__ A,
                 const _Float16* __restrict__ Bt,
                 const float* __restrict__ bias,
                 const float* __restrict__ resid,
                 float* __restrict__ Cf,
                 _Float16* __restrict__ Ch,
                 int M, int Nc, int K, int lda, int ldb,
                 float scale, int relu,
                 long sA, long sB, long sCf, long sCh, long sR) {
    int lane = threadIdx.x & 31, wv = threadIdx.x >> 5;
    long z = blockIdx.z;
    A  += z * sA;
    Bt += z * sB;
    long rowBase = ((long)blockIdx.x * 4 + wv) * (16 * RT);
    long colBase = (long)blockIdx.y * 64;

    v8f acc[RT][4];
#pragma unroll
    for (int i = 0; i < RT; ++i)
#pragma unroll
        for (int t = 0; t < 4; ++t) acc[i][t] = vzero8();

    v16h a[RT], b[4];
#pragma unroll
    for (int i = 0; i < RT; ++i) a[i] = load_afrag(A, rowBase + 16 * i, lda, 0, lane);
#pragma unroll
    for (int t = 0; t < 4; ++t)  b[t] = load_bfrag(Bt, colBase + 16 * t, ldb, 0, lane);

    for (int k = 0; k < K; k += 32) {
        v16h an[RT], bn[4];
        int kn = k + 32;
        if (kn < K) {
            if (kn + 32 < K)
                __builtin_prefetch((const void*)(A + (rowBase + (lane & 15)) * (long)lda
                                                 + kn + 32), 0, 3);
#pragma unroll
            for (int i = 0; i < RT; ++i)
                an[i] = load_afrag(A, rowBase + 16 * i, lda, kn, lane);
#pragma unroll
            for (int t = 0; t < 4; ++t)
                bn[t] = load_bfrag(Bt, colBase + 16 * t, ldb, kn, lane);
        }
#pragma unroll
        for (int i = 0; i < RT; ++i)
#pragma unroll
            for (int t = 0; t < 4; ++t)
                acc[i][t] = wmma_f16(a[i], b[t], acc[i][t]);
#pragma unroll
        for (int i = 0; i < RT; ++i) a[i] = an[i];
#pragma unroll
        for (int t = 0; t < 4; ++t)  b[t] = bn[t];
    }

    int half = lane >> 4, coll = lane & 15;
    if (Cf)    Cf    += z * sCf;
    if (Ch)    Ch    += z * sCh;
    if (resid) resid += z * sR;
#pragma unroll
    for (int i = 0; i < RT; ++i) {
#pragma unroll
        for (int t = 0; t < 4; ++t) {
            long col = colBase + t * 16 + coll;
            float bv = bias ? bias[col] : 0.0f;
#pragma unroll
            for (int r = 0; r < 8; ++r) {
                long row = rowBase + 16 * i + r + 8 * half;  // C: M = r + 8*(lane/16)
                float v = acc[i][t][r] * scale + bv;
                if (relu) v = fmaxf(v, 0.0f);
                long idx = row * (long)Nc + col;
                if (resid) v += resid[idx];
                if (Cf) Cf[idx] = v;
                if (Ch) Ch[idx] = (_Float16)v;
            }
        }
    }
}

// ---------------- Flash attention ------------------------------------------
// Q, K: [H, B*N, DK] f16 (Q pre-scaled by 1/sqrt(DK)); Vt: [H, DK, B*N] f16.
// heads out: [B*N, H*DK] f16. grid = (N/64, B, H), block = 128 (4 waves, each
// wave owns a 16-query tile and streams 32 keys per iteration).
__global__ void __launch_bounds__(128, 1)
flash_attn_kernel(const _Float16* __restrict__ Q,
                  const _Float16* __restrict__ Km,
                  const _Float16* __restrict__ Vt,
                  _Float16* __restrict__ heads) {
    __shared__ _Float16 pbuf[4 * 16 * 32];   // per-wave 16x32 P staging tile
    int lane = threadIdx.x & 31, wv = threadIdx.x >> 5;
    int half = lane >> 4, coll = lane & 15;
    int b = blockIdx.y, h = blockIdx.z;
    int qBase = (blockIdx.x * 4 + wv) * 16;

    const _Float16* Qh = Q  + (long)h * BNt * DKK;
    const _Float16* Kh = Km + (long)h * BNt * DKK;
    const _Float16* Vh = Vt + (long)h * DKK * BNt;
    long tokQ = (long)b * NNq + qBase;

    v16h aq0 = load_afrag(Qh, tokQ, DKK, 0,  lane);   // Q tile, d = 0..31
    v16h aq1 = load_afrag(Qh, tokQ, DKK, 32, lane);   // Q tile, d = 32..63

    v8f o[4];
    float m[8], l[8];
#pragma unroll
    for (int t = 0; t < 4; ++t) o[t] = vzero8();
#pragma unroll
    for (int r = 0; r < 8; ++r) { m[r] = -1e30f; l[r] = 0.0f; }

    _Float16* myp = pbuf + wv * (16 * 32);

    for (int c = 0; c < NNq / 32; ++c) {
        long kvTok = (long)b * NNq + c * 32;

        // S = Q K^T for 16 queries x 32 keys (two 16x16 C tiles).
        // K fragments consumed in pairs to keep live ranges short.
        v8f s0 = vzero8(), s1 = vzero8();
        {
            v16h bk0a = load_bfrag(Kh, kvTok, DKK, 0,  lane);
            v16h bk0b = load_bfrag(Kh, kvTok, DKK, 32, lane);
            s0 = wmma_f16(aq0, bk0a, s0);
            s0 = wmma_f16(aq1, bk0b, s0);
        }
        {
            v16h bk1a = load_bfrag(Kh, kvTok + 16, DKK, 0,  lane);
            v16h bk1b = load_bfrag(Kh, kvTok + 16, DKK, 32, lane);
            s1 = wmma_f16(aq0, bk1a, s1);
            s1 = wmma_f16(aq1, bk1b, s1);
        }

        // Issue V fragments now: they cover the whole softmax VALU section and
        // the LDS P bounce, but aren't live across the K fragments.
        v16h bv[4];
#pragma unroll
        for (int t = 0; t < 4; ++t)
            bv[t] = load_bfrag(Vh, t * 16, BNt, kvTok, lane);

        // online softmax; rows live across 16-lane groups -> xor reductions
#pragma unroll
        for (int r = 0; r < 8; ++r) {
            float v0 = s0[r], v1 = s1[r];
            float mx = fmaxf(v0, v1);
#pragma unroll
            for (int d = 8; d >= 1; d >>= 1) mx = fmaxf(mx, __shfl_xor(mx, d, 32));
            float mnew  = fmaxf(m[r], mx);
            float alpha = __expf(m[r] - mnew);
            float p0 = __expf(v0 - mnew);
            float p1 = __expf(v1 - mnew);
            float sm = p0 + p1;
#pragma unroll
            for (int d = 8; d >= 1; d >>= 1) sm += __shfl_xor(sm, d, 32);
            l[r] = l[r] * alpha + sm;
            m[r] = mnew;
#pragma unroll
            for (int t = 0; t < 4; ++t) o[t][r] *= alpha;
            int prow = r + 8 * half;                  // C-layout row
            myp[prow * 32 + coll]      = (_Float16)p0;
            myp[prow * 32 + 16 + coll] = (_Float16)p1;
        }

        asm volatile("s_wait_dscnt 0" ::: "memory");  // LDS store->load fence

        // reload P as a 16x32 A-fragment
        v8h plo = *(const v8h*)(myp + coll * 32 + 8 * half);
        v8h phi = *(const v8h*)(myp + coll * 32 + 16 + 8 * half);
        v16h ap;
#pragma unroll
        for (int i = 0; i < 8; ++i) { ap[i] = plo[i]; ap[i + 8] = phi[i]; }

        // O += P @ V  (V^T is row-contiguous in kv -> clean B fragments)
#pragma unroll
        for (int t = 0; t < 4; ++t)
            o[t] = wmma_f16(ap, bv[t], o[t]);
    }

#pragma unroll
    for (int r = 0; r < 8; ++r) {
        float inv = 1.0f / l[r];
        long row = (long)b * NNq + qBase + r + 8 * half;
#pragma unroll
        for (int t = 0; t < 4; ++t)
            heads[row * (HHn * DKK) + h * DKK + t * 16 + coll] =
                (_Float16)(o[t][r] * inv);
    }
}

// ---------------- BatchNorm (training-mode stats, biased var) ---------------
__global__ void bn_stats_kernel(const float* __restrict__ s,
                                float* __restrict__ mu, float* __restrict__ var,
                                int Mrows, int Dfeat) {
    __shared__ float ss[256], sq[256];
    int f = blockIdx.x, tid = threadIdx.x;
    float a = 0.f, b = 0.f;
    for (int r = tid; r < Mrows; r += blockDim.x) {
        float v = s[(long)r * Dfeat + f];
        a += v; b += v * v;
    }
    ss[tid] = a; sq[tid] = b;
    __syncthreads();
    for (int off = 128; off > 0; off >>= 1) {
        if (tid < off) { ss[tid] += ss[tid + off]; sq[tid] += sq[tid + off]; }
        __syncthreads();
    }
    if (tid == 0) {
        float mean = ss[0] / Mrows;
        mu[f]  = mean;
        var[f] = sq[0] / Mrows - mean * mean;
    }
}

__global__ void bn_norm_kernel(const float* __restrict__ s,
                               const float* __restrict__ mu,
                               const float* __restrict__ var,
                               const float* __restrict__ w,
                               const float* __restrict__ bia,
                               float* __restrict__ yf, _Float16* __restrict__ yh,
                               long total, int Dfeat) {
    long i = (long)blockIdx.x * blockDim.x + threadIdx.x;
    if (i >= total) return;
    int f = (int)(i % Dfeat);
    float v = (s[i] - mu[f]) * rsqrtf(var[f] + EPSb) * w[f] + bia[f];
    if (yf) yf[i] = v;
    if (yh) yh[i] = (_Float16)v;
}

// ---------------- Layout prep ----------------------------------------------
__global__ void cvt_f32_f16_kernel(const float* __restrict__ in,
                                   _Float16* __restrict__ out, long n) {
    long i = (long)blockIdx.x * blockDim.x + threadIdx.x;
    if (i < n) out[i] = (_Float16)in[i];
}

// in [z][R][C] f32 -> out [z][C][R] f16
__global__ void transpose_cvt_kernel(const float* __restrict__ in,
                                     _Float16* __restrict__ out, int R, int C) {
    long base = (long)blockIdx.z * R * C;
    long i = (long)blockIdx.x * blockDim.x + threadIdx.x;
    if (i >= (long)R * C) return;
    int r = (int)(i / C), c = (int)(i % C);
    out[base + (long)c * R + r] = (_Float16)in[base + i];
}

// ---------------- Host orchestration ---------------------------------------
extern "C" void kernel_launch(void* const* d_in, const int* in_sizes, int n_in,
                              void* d_out, int out_size, void* d_ws, size_t ws_size,
                              hipStream_t stream) {
    (void)in_sizes; (void)n_in; (void)out_size; (void)ws_size;
    const float* x     = (const float*)d_in[0];   // [B,N,D]
    const float* Wq    = (const float*)d_in[1];   // [H,D,DK]
    const float* Wk    = (const float*)d_in[2];
    const float* Wv    = (const float*)d_in[3];
    const float* Wo    = (const float*)d_in[4];   // [H,DK,D] == flat [H*DK, D]
    const float* bn1w  = (const float*)d_in[5];
    const float* bn1b  = (const float*)d_in[6];
    const float* W1    = (const float*)d_in[7];   // [D,FF]
    const float* b1    = (const float*)d_in[8];
    const float* W2    = (const float*)d_in[9];   // [FF,D]
    const float* b2    = (const float*)d_in[10];
    const float* bn2w  = (const float*)d_in[11];
    const float* bn2b  = (const float*)d_in[12];
    float* out = (float*)d_out;

    char* wp = (char*)d_ws;
    auto alloc = [&](size_t bytes) -> char* {
        char* p = wp;
        wp += (bytes + 255) & ~(size_t)255;
        return p;
    };
    _Float16* xh    = (_Float16*)alloc((size_t)BNt * DDm * 2);
    _Float16* Wqt   = (_Float16*)alloc((size_t)HHn * DKK * DDm * 2);
    _Float16* Wkt   = (_Float16*)alloc((size_t)HHn * DKK * DDm * 2);
    _Float16* Wvt   = (_Float16*)alloc((size_t)HHn * DKK * DDm * 2);
    _Float16* Wot   = (_Float16*)alloc((size_t)DDm * HHn * DKK * 2);
    _Float16* W1t   = (_Float16*)alloc((size_t)FFn * DDm * 2);
    _Float16* W2t   = (_Float16*)alloc((size_t)DDm * FFn * 2);
    _Float16* Qm    = (_Float16*)alloc((size_t)HHn * BNt * DKK * 2);
    _Float16* Kmb   = (_Float16*)alloc((size_t)HHn * BNt * DKK * 2);
    _Float16* Vtb   = (_Float16*)alloc((size_t)HHn * DKK * BNt * 2);
    _Float16* heads = (_Float16*)alloc((size_t)BNt * HHn * DKK * 2);
    float*    s1    = (float*)   alloc((size_t)BNt * DDm * 4);
    float*    y1    = (float*)   alloc((size_t)BNt * DDm * 4);
    _Float16* y1h   = (_Float16*)alloc((size_t)BNt * DDm * 2);
    _Float16* h1    = (_Float16*)alloc((size_t)BNt * FFn * 2);
    float*    s2    = (float*)   alloc((size_t)BNt * DDm * 4);
    float*    mu1   = (float*)   alloc((size_t)DDm * 4);
    float*    var1  = (float*)   alloc((size_t)DDm * 4);
    float*    mu2   = (float*)   alloc((size_t)DDm * 4);
    float*    var2  = (float*)   alloc((size_t)DDm * 4);

    const long nXD = (long)BNt * DDm;            // 4,194,304
    const float invSqrtDK = 0.125f;              // 1/sqrt(64)

    // -- layout prep --
    cvt_f32_f16_kernel<<<(nXD + 255) / 256, 256, 0, stream>>>(x, xh, nXD);
    transpose_cvt_kernel<<<dim3((DDm * DKK) / 256, 1, HHn), 256, 0, stream>>>(Wq, Wqt, DDm, DKK);
    transpose_cvt_kernel<<<dim3((DDm * DKK) / 256, 1, HHn), 256, 0, stream>>>(Wk, Wkt, DDm, DKK);
    transpose_cvt_kernel<<<dim3((DDm * DKK) / 256, 1, HHn), 256, 0, stream>>>(Wv, Wvt, DDm, DKK);
    transpose_cvt_kernel<<<dim3((HHn * DKK * DDm) / 256, 1, 1), 256, 0, stream>>>(Wo, Wot, HHn * DKK, DDm);
    transpose_cvt_kernel<<<dim3((DDm * FFn) / 256, 1, 1), 256, 0, stream>>>(W1, W1t, DDm, FFn);
    transpose_cvt_kernel<<<dim3((FFn * DDm) / 256, 1, 1), 256, 0, stream>>>(W2, W2t, FFn, DDm);

    // -- QKV projections (per head, batched over grid.z) --
    // Q[h] = (x @ Wq[h]) * 1/sqrt(DK)  -> [H, BN, DK] f16
    gemm_wmma_kernel<2><<<dim3(BNt / 128, DKK / 64, HHn), 128, 0, stream>>>(
        xh, Wqt, nullptr, nullptr, nullptr, Qm,
        BNt, DKK, DDm, DDm, DDm, invSqrtDK, 0,
        0, (long)DKK * DDm, 0, (long)BNt * DKK, 0);
    gemm_wmma_kernel<2><<<dim3(BNt / 128, DKK / 64, HHn), 128, 0, stream>>>(
        xh, Wkt, nullptr, nullptr, nullptr, Kmb,
        BNt, DKK, DDm, DDm, DDm, 1.0f, 0,
        0, (long)DKK * DDm, 0, (long)BNt * DKK, 0);
    // Vt[h] = Wv[h]^T @ x^T -> [H, DK, BN] f16 (transposed V for PV B-frags)
    gemm_wmma_kernel<1><<<dim3(DKK / 64, BNt / 64, HHn), 128, 0, stream>>>(
        Wvt, xh, nullptr, nullptr, nullptr, Vtb,
        DKK, BNt, DDm, DDm, DDm, 1.0f, 0,
        (long)DKK * DDm, 0, 0, (long)DKK * BNt, 0);

    // -- attention (flash, online softmax) -> heads [BN, H*DK] f16 --
    flash_attn_kernel<<<dim3(NNq / 64, BB, HHn), 128, 0, stream>>>(Qm, Kmb, Vtb, heads);

    // -- output projection + residual: s1 = heads @ WoFlat + x --
    gemm_wmma_kernel<2><<<dim3(BNt / 128, DDm / 64, 1), 128, 0, stream>>>(
        heads, Wot, nullptr, x, s1, nullptr,
        BNt, DDm, HHn * DKK, HHn * DKK, DDm, 1.0f, 0, 0, 0, 0, 0, 0);

    // -- batchnorm 1 --
    bn_stats_kernel<<<DDm, 256, 0, stream>>>(s1, mu1, var1, BNt, DDm);
    bn_norm_kernel<<<(nXD + 255) / 256, 256, 0, stream>>>(
        s1, mu1, var1, bn1w, bn1b, y1, y1h, nXD, DDm);

    // -- FFN: h1 = relu(y1 @ W1 + b1); s2 = h1 @ W2 + b2 + y1 --
    gemm_wmma_kernel<2><<<dim3(BNt / 128, FFn / 64, 1), 128, 0, stream>>>(
        y1h, W1t, b1, nullptr, nullptr, h1,
        BNt, FFn, DDm, DDm, DDm, 1.0f, 1, 0, 0, 0, 0, 0);
    gemm_wmma_kernel<2><<<dim3(BNt / 128, DDm / 64, 1), 128, 0, stream>>>(
        h1, W2t, b2, y1, s2, nullptr,
        BNt, DDm, FFn, FFn, FFn, 1.0f, 0, 0, 0, 0, 0, 0);

    // -- batchnorm 2 -> final output (f32) --
    bn_stats_kernel<<<DDm, 256, 0, stream>>>(s2, mu2, var2, BNt, DDm);
    bn_norm_kernel<<<(nXD + 255) / 256, 256, 0, stream>>>(
        s2, mu2, var2, bn2w, bn2b, out, nullptr, nXD, DDm);
}
